// MINDConv_12567074308479
// MI455X (gfx1250) — compile-verified, matrix operand order checked by reading, not run on hardware
//
#include <hip/hip_runtime.h>

// -------------------- CDNA5 WMMA types / helpers --------------------
typedef __attribute__((ext_vector_type(16))) __bf16 v16bf;
typedef __attribute__((ext_vector_type(8)))  float  v8f;
typedef int v4i_vec __attribute__((vector_size(16)));
typedef __attribute__((address_space(1))) v4i_vec* gv4i_ptr;   // global v4i*
typedef __attribute__((address_space(3))) v4i_vec* lv4i_ptr;   // LDS v4i*

#define WMMA_BF16(A, B, C) \
  __builtin_amdgcn_wmma_f32_16x16x32_bf16(false, (A), false, (B), (short)0, (C), false, false)

#if __has_builtin(__builtin_amdgcn_global_load_async_to_lds_b128)
#define HAVE_ASYNC_LDS 1
#else
#define HAVE_ASYNC_LDS 0
#endif

__device__ __forceinline__ float leaky02(float x)  { return x > 0.0f ? x : 0.2f * x; }
__device__ __forceinline__ float sigmoidf_(float x){ return 1.0f / (1.0f + __expf(-x)); }
// intra-wave LDS RAW fence (wave32: DS ops are in-order per wave; wait for returns)
__device__ __forceinline__ void lds_fence() { __asm__ volatile("s_wait_dscnt 0" ::: "memory"); }
__device__ __forceinline__ void wait_async0() {
#if __has_builtin(__builtin_amdgcn_s_wait_asynccnt)
  __builtin_amdgcn_s_wait_asynccnt(0);
#else
  __asm__ volatile("s_wait_asynccnt 0" ::: "memory");
#endif
}

// -------------------- weight pre-swizzle into WMMA B-fragment order --------------------
// B fragment (K=32 x N=16, bf16): lane<16 holds n=lane, K={0..7,16..23}; lane>=16 holds
// n=lane-16, K={8..15,24..31}; element j -> k = (j<8 ? j : j+8) + (lane>=16 ? 8 : 0).
// Fragment storage: [(kt*nTiles+nt)*32 + lane]*16 + j  (contiguous 32B per lane).
__global__ void pack_b_kernel(const float* __restrict__ W, __bf16* __restrict__ outp,
                              int Ktot, int Ntot, int kTiles, int nTiles) {
  int idx = blockIdx.x * blockDim.x + threadIdx.x;
  int total = kTiles * nTiles * 32 * 16;
  if (idx >= total) return;
  int j    = idx & 15;
  int lane = (idx >> 4) & 31;
  int t    = idx >> 9;
  int nt   = t % nTiles;
  int kt   = t / nTiles;
  int n = (lane & 15) + nt * 16;
  int k = ((j < 8) ? j : j + 8) + ((lane & 16) ? 8 : 0) + kt * 32;
  float v = (n < Ntot && k < Ktot) ? W[k * Ntot + n] : 0.0f;
  outp[idx] = (__bf16)v;
}

// -------------------- node kernel: g_src, g_dst, a_dst --------------------
// One wave per 16-node tile. A fragment (16x32 bf16): lane<16 holds row=lane,
// K={0..7,16..23}; lane>=16 row=lane-16, K={8..15,24..31}.
// C/D fragment (16x16 f32): vgpr r -> m = r + (lane>=16?8:0), n = lane&15.
__global__ __launch_bounds__(256) void node_kernel(
    const float* __restrict__ h,
    const __bf16* __restrict__ pWsrc, const float* __restrict__ bsrc,
    const __bf16* __restrict__ pWdst, const float* __restrict__ bdst,
    const __bf16* __restrict__ pWa1,  const float* __restrict__ ba1,
    const __bf16* __restrict__ pWa2,  const float* __restrict__ ba2,
    float* __restrict__ g_src, float* __restrict__ g_dst, float* __restrict__ a_dst,
    int N)
{
  __shared__ float smem[8][16 * 68];   // stride 68: 16B-aligned rows, conflict-free banks
  const int lane = (int)(threadIdx.x & 31u);
  const int wid  = (int)(threadIdx.x >> 5);
  const int tile = (int)blockIdx.x * 8 + wid;
  const int m0   = tile * 16;
  if (m0 >= N) return;
  const bool full = (m0 + 16 <= N);    // wave-uniform -> single scalar branch per tile

  float (*sm)[68] = reinterpret_cast<float (*)[68]>(&smem[wid][0]);

  const int mrow = lane & 15;
  const int hi   = (lane & 16) ? 1 : 0;
  const int kb   = hi ? 8 : 0;

  __builtin_prefetch((const void*)pWsrc, 0, 0);
  __builtin_prefetch((const void*)pWdst, 0, 0);

  const int rowA = m0 + mrow;
  const int rAc  = rowA < N ? rowA : (N - 1);
  const float* hrow = h + (size_t)rAc * 64;

  // A fragments of h (two K-tiles covering F=64)
  v16bf a0, a1;
#pragma unroll
  for (int j = 0; j < 8; ++j) {
    a0[j]     = (__bf16)hrow[kb + j];
    a0[8 + j] = (__bf16)hrow[kb + 16 + j];
    a1[j]     = (__bf16)hrow[32 + kb + j];
    a1[8 + j] = (__bf16)hrow[32 + kb + 16 + j];
  }

  // ---- g_src = h @ W_src + b_src ----
#pragma unroll
  for (int nt = 0; nt < 4; ++nt) {
    v8f acc = {};
    v16bf b0 = *(const v16bf*)(pWsrc + ((size_t)((0 * 4 + nt) * 32 + lane)) * 16);
    v16bf b1 = *(const v16bf*)(pWsrc + ((size_t)((1 * 4 + nt) * 32 + lane)) * 16);
    acc = WMMA_BF16(a0, b0, acc);
    acc = WMMA_BF16(a1, b1, acc);
    const int n = mrow + nt * 16;
    const float bias = bsrc[n];
    float* gp = g_src + (size_t)(m0 + hi * 8) * 64 + n;
    if (full) {
#pragma unroll
      for (int r = 0; r < 8; ++r) gp[(size_t)r * 64] = acc[r] + bias;
    } else {
#pragma unroll
      for (int r = 0; r < 8; ++r)
        if (m0 + r + hi * 8 < N) gp[(size_t)r * 64] = acc[r] + bias;
    }
  }

  // ---- g_dst = h @ W_dst + b_dst (stage to LDS for the a_dst MLP) ----
#pragma unroll
  for (int nt = 0; nt < 4; ++nt) {
    v8f acc = {};
    v16bf b0 = *(const v16bf*)(pWdst + ((size_t)((0 * 4 + nt) * 32 + lane)) * 16);
    v16bf b1 = *(const v16bf*)(pWdst + ((size_t)((1 * 4 + nt) * 32 + lane)) * 16);
    acc = WMMA_BF16(a0, b0, acc);
    acc = WMMA_BF16(a1, b1, acc);
    const int n = mrow + nt * 16;
    const float bias = bdst[n];
    float* gp = g_dst + (size_t)(m0 + hi * 8) * 64 + n;
    if (full) {
#pragma unroll
      for (int r = 0; r < 8; ++r) {
        const float v = acc[r] + bias;
        sm[r + hi * 8][n] = v;
        gp[(size_t)r * 64] = v;
      }
    } else {
#pragma unroll
      for (int r = 0; r < 8; ++r) {
        const float v = acc[r] + bias;
        sm[r + hi * 8][n] = v;
        if (m0 + r + hi * 8 < N) gp[(size_t)r * 64] = v;
      }
    }
  }

  lds_fence();

  // ---- a_dst MLP: leaky -> (64x32) -> leaky -> (32x8) -> sigmoid ----
  v16bf d0, d1;
#pragma unroll
  for (int j = 0; j < 8; ++j) {
    d0[j]     = (__bf16)leaky02(sm[mrow][kb + j]);
    d0[8 + j] = (__bf16)leaky02(sm[mrow][kb + 16 + j]);
    d1[j]     = (__bf16)leaky02(sm[mrow][32 + kb + j]);
    d1[8 + j] = (__bf16)leaky02(sm[mrow][32 + kb + 16 + j]);
  }

#pragma unroll
  for (int nt = 0; nt < 2; ++nt) {
    v8f acc = {};
    v16bf b0 = *(const v16bf*)(pWa1 + ((size_t)((0 * 2 + nt) * 32 + lane)) * 16);
    v16bf b1 = *(const v16bf*)(pWa1 + ((size_t)((1 * 2 + nt) * 32 + lane)) * 16);
    acc = WMMA_BF16(d0, b0, acc);
    acc = WMMA_BF16(d1, b1, acc);
    const int n = mrow + nt * 16;
    const float bias = ba1[n];
#pragma unroll
    for (int r = 0; r < 8; ++r)
      sm[r + hi * 8][n] = leaky02(acc[r] + bias);   // reuse LDS (cols 0..31)
  }

  lds_fence();

  v16bf e2;
#pragma unroll
  for (int j = 0; j < 8; ++j) {
    e2[j]     = (__bf16)sm[mrow][kb + j];
    e2[8 + j] = (__bf16)sm[mrow][kb + 16 + j];
  }
  {
    v8f acc = {};
    v16bf b0 = *(const v16bf*)(pWa2 + (size_t)lane * 16);
    acc = WMMA_BF16(e2, b0, acc);
    if (mrow < 8) {
      const float bias = ba2[mrow];
      float* ap = a_dst + (size_t)(m0 + hi * 8) * 8 + mrow;
      if (full) {
#pragma unroll
        for (int r = 0; r < 8; ++r) ap[(size_t)r * 8] = sigmoidf_(acc[r] + bias);
      } else {
#pragma unroll
        for (int r = 0; r < 8; ++r)
          if (m0 + r + hi * 8 < N) ap[(size_t)r * 8] = sigmoidf_(acc[r] + bias);
      }
    }
  }
}

// -------------------- seed output: h_next = a_dst * g_dst (+ atomic agg later) ----------
__global__ void init_out_kernel(const float* __restrict__ g_dst, const float* __restrict__ a_dst,
                                float* __restrict__ out, size_t total) {
  size_t i = (size_t)blockIdx.x * blockDim.x + threadIdx.x;
  if (i >= total) return;
  const size_t n = i >> 6;
  const int   f  = (int)(i & 63);
  out[i] = a_dst[n * 8 + (f >> 3)] * g_dst[i];
}

// -------------------- edge kernel: attn MLP + scatter --------------------
// edgeW = [Wa1_src fragments : 4*512 bf16][Wa2_src fragments : 512 bf16]  (contiguous, 5120B)
__global__ __launch_bounds__(256) void edge_kernel(
    const float* __restrict__ g_src, const float* __restrict__ g_dst,
    const __bf16* __restrict__ edgeW,
    const float* __restrict__ ba1, const float* __restrict__ ba2,
    const int* __restrict__ esrc, const int* __restrict__ edst,
    float* __restrict__ out, int E)
{
  __shared__ float ymem[8][16 * 36];   // stride 36: conflict-free for 16-row reads
  __shared__ float amem[8][16 * 8];
  __shared__ int   simem[8][16];
  __shared__ int   dimem[8][16];
  __shared__ __align__(16) __bf16 wsh[5 * 512];   // staged MLP weights (5120 B)

  const int tid  = (int)threadIdx.x;
  const int lane = tid & 31;
  const int wid  = tid >> 5;

  // ---- stage the (tiny, L2-hot) MLP weights into LDS once per block ----
#if HAVE_ASYNC_LDS
  {
    const char* gsrcb = (const char*)edgeW;
    char*       ldsb  = (char*)wsh;
#pragma unroll
    for (int i = tid; i < 320; i += 256) {   // 320 x 16B = 5120B
      __builtin_amdgcn_global_load_async_to_lds_b128(
          (gv4i_ptr)(gsrcb + (size_t)i * 16),
          (lv4i_ptr)(ldsb + (size_t)i * 16), 0, 0);
    }
    wait_async0();
  }
#else
  {
    const uint4* gs16 = (const uint4*)edgeW;
    uint4*       ls16 = (uint4*)wsh;
    for (int i = tid; i < 320; i += 256) ls16[i] = gs16[i];
  }
#endif
  __syncthreads();

  const int tile = (int)blockIdx.x * 8 + wid;
  const int e0   = tile * 16;
  if (e0 >= E) return;
  const bool fullE = (e0 + 16 <= E);   // wave-uniform

  float (*ys)[36] = reinterpret_cast<float (*)[36]>(&ymem[wid][0]);
  float (*as)[8]  = reinterpret_cast<float (*)[8]>(&amem[wid][0]);
  int* si = simem[wid];
  int* di = dimem[wid];
  const __bf16* wa1 = wsh;
  const __bf16* wa2 = wsh + 4 * 512;

  const int mrow = lane & 15;
  const int hi   = (lane & 16) ? 1 : 0;
  const int kb   = hi ? 8 : 0;

  const int er = e0 + mrow;
  int s, dd;
  if (fullE || er < E) { s = esrc[er]; dd = edst[er]; }
  else                 { s = 0; dd = -1; }
  if (lane < 16) { si[lane] = s; di[lane] = dd; }
  const int ddc = dd < 0 ? 0 : dd;

  const float* gs = g_src + (size_t)s * 64;     // L2-resident (51MB << 192MB L2)
  const float* gd = g_dst + (size_t)ddc * 64;

  // msg = g_src[src] + g_dst[dst]; A fragments of leaky(msg)
  v16bf a0, a1;
#pragma unroll
  for (int j = 0; j < 8; ++j) {
    a0[j]     = (__bf16)leaky02(gs[kb + j]           + gd[kb + j]);
    a0[8 + j] = (__bf16)leaky02(gs[kb + 16 + j]      + gd[kb + 16 + j]);
    a1[j]     = (__bf16)leaky02(gs[32 + kb + j]      + gd[32 + kb + j]);
    a1[8 + j] = (__bf16)leaky02(gs[32 + kb + 16 + j] + gd[32 + kb + 16 + j]);
  }

  // layer1: (16x64) @ (64x32)
#pragma unroll
  for (int nt = 0; nt < 2; ++nt) {
    v8f acc = {};
    v16bf b0 = *(const v16bf*)(wa1 + ((size_t)((0 * 2 + nt) * 32 + lane)) * 16);
    v16bf b1 = *(const v16bf*)(wa1 + ((size_t)((1 * 2 + nt) * 32 + lane)) * 16);
    acc = WMMA_BF16(a0, b0, acc);
    acc = WMMA_BF16(a1, b1, acc);
    const int n = mrow + nt * 16;
    const float bias = ba1[n];
#pragma unroll
    for (int r = 0; r < 8; ++r)
      ys[r + hi * 8][n] = leaky02(acc[r] + bias);
  }

  lds_fence();

  // layer2: (16x32) @ (32x16-padded), sigmoid -> a_src in LDS
  v16bf e2;
#pragma unroll
  for (int j = 0; j < 8; ++j) {
    e2[j]     = (__bf16)ys[mrow][kb + j];
    e2[8 + j] = (__bf16)ys[mrow][kb + 16 + j];
  }
  {
    v8f acc = {};
    v16bf b0 = *(const v16bf*)(wa2 + (size_t)lane * 16);
    acc = WMMA_BF16(e2, b0, acc);
    if (mrow < 8) {
      const float bias = ba2[mrow];
#pragma unroll
      for (int r = 0; r < 8; ++r)
        as[r + hi * 8][mrow] = sigmoidf_(acc[r] + bias);
    }
  }

  lds_fence();

  // scatter: out[dst] += a_src[h] * g_src[src][h*8+d]  (2 coalesced cols/lane x 16 edges)
  if (fullE) {
#pragma unroll 4
    for (int m = 0; m < 16; ++m) {
      const float* gsm = g_src + (size_t)si[m] * 64;
      float* od = out + (size_t)di[m] * 64;
#pragma unroll
      for (int half = 0; half < 2; ++half) {
        const int col = lane + 32 * half;
        atomicAdd(od + col, as[m][col >> 3] * gsm[col]);
      }
    }
  } else {
    for (int m = 0; m < 16; ++m) {
      const int dm = di[m];
      if (dm < 0) continue;
      const float* gsm = g_src + (size_t)si[m] * 64;
      float* od = out + (size_t)dm * 64;
#pragma unroll
      for (int half = 0; half < 2; ++half) {
        const int col = lane + 32 * half;
        atomicAdd(od + col, as[m][col >> 3] * gsm[col]);
      }
    }
  }
}

// -------------------- host launch --------------------
extern "C" void kernel_launch(void* const* d_in, const int* in_sizes, int n_in,
                              void* d_out, int out_size, void* d_ws, size_t ws_size,
                              hipStream_t stream) {
  const float* h     = (const float*)d_in[0];
  const float* W_src = (const float*)d_in[1];
  const float* b_src = (const float*)d_in[2];
  const float* W_dst = (const float*)d_in[3];
  const float* b_dst = (const float*)d_in[4];
  const float* Wa1s  = (const float*)d_in[5];
  const float* ba1s  = (const float*)d_in[6];
  const float* Wa2s  = (const float*)d_in[7];
  const float* ba2s  = (const float*)d_in[8];
  const float* Wa1d  = (const float*)d_in[9];
  const float* ba1d  = (const float*)d_in[10];
  const float* Wa2d  = (const float*)d_in[11];
  const float* ba2d  = (const float*)d_in[12];
  const int*   eidx  = (const int*)d_in[13];   // int32 (JAX w/o x64 downgrades int64)

  const int N = in_sizes[0] / 64;
  const int E = in_sizes[13] / 2;
  (void)n_in; (void)out_size; (void)ws_size;

  char* wp = (char*)d_ws;
  auto carve = [&](size_t bytes) -> void* {
    void* p = (void*)wp;
    wp += (bytes + 255) & ~(size_t)255;
    return p;
  };
  float*  g_src = (float*)carve((size_t)N * 64 * sizeof(float));
  float*  g_dst = (float*)carve((size_t)N * 64 * sizeof(float));
  float*  a_dst = (float*)carve((size_t)N * 8 * sizeof(float));
  __bf16* pWsrc = (__bf16*)carve((size_t)8 * 512 * 2);
  __bf16* pWdst = (__bf16*)carve((size_t)8 * 512 * 2);
  __bf16* pWa1d = (__bf16*)carve((size_t)4 * 512 * 2);
  __bf16* pWa2d = (__bf16*)carve((size_t)1 * 512 * 2);
  // edge weights carved contiguously: [Wa1_src frags 4096B][Wa2_src frags 1024B]
  __bf16* pWa1s = (__bf16*)carve((size_t)4 * 512 * 2);   // 4096B (multiple of 256)
  __bf16* pWa2s = (__bf16*)carve((size_t)1 * 512 * 2);   // contiguous after pWa1s

  // weight pre-swizzle (tiny)
  pack_b_kernel<<<dim3(16), dim3(256), 0, stream>>>(W_src, pWsrc, 64, 64, 2, 4);
  pack_b_kernel<<<dim3(16), dim3(256), 0, stream>>>(W_dst, pWdst, 64, 64, 2, 4);
  pack_b_kernel<<<dim3(8),  dim3(256), 0, stream>>>(Wa1s,  pWa1s, 64, 32, 2, 2);
  pack_b_kernel<<<dim3(8),  dim3(256), 0, stream>>>(Wa1d,  pWa1d, 64, 32, 2, 2);
  pack_b_kernel<<<dim3(2),  dim3(256), 0, stream>>>(Wa2s,  pWa2s, 32, 8, 1, 1);
  pack_b_kernel<<<dim3(2),  dim3(256), 0, stream>>>(Wa2d,  pWa2d, 32, 8, 1, 1);

  const int nodeTiles = (N + 15) / 16;
  node_kernel<<<dim3((unsigned)((nodeTiles + 7) / 8)), dim3(256), 0, stream>>>(
      h, pWsrc, b_src, pWdst, b_dst, pWa1d, ba1d, pWa2d, ba2d,
      g_src, g_dst, a_dst, N);

  const size_t total = (size_t)N * 64;
  init_out_kernel<<<dim3((unsigned)((total + 255) / 256)), dim3(256), 0, stream>>>(
      g_dst, a_dst, (float*)d_out, total);

  const int edgeTiles = (E + 15) / 16;
  edge_kernel<<<dim3((unsigned)((edgeTiles + 7) / 8)), dim3(256), 0, stream>>>(
      g_src, g_dst, pWa1s, ba1s, ba2s, eidx, eidx + E, (float*)d_out, E);
}
